// CausalMultiheadAttention_51153060495412
// MI455X (gfx1250) — matrix operand order repaired and checked by reading, not compiled, not run on hardware
//
#include <hip/hip_runtime.h>
#include <hip/hip_bf16.h>

#define TSEQ   2048
#define NH     16
#define HD     64
#define DMODEL 1024
#define BB     4
#define MTOT   (BB * TSEQ)   // 8192 tokens

typedef __bf16 bf16_t;
typedef bf16_t v16bf __attribute__((ext_vector_type(16)));
typedef bf16_t v8bf  __attribute__((ext_vector_type(8)));
typedef float  v8f   __attribute__((ext_vector_type(8)));

static __device__ __forceinline__ bf16_t f2bf(float f) { return (bf16_t)f; }

static __device__ __forceinline__ v8bf ld8(const bf16_t* p) {
  return *reinterpret_cast<const v8bf*>(p);
}

static __device__ __forceinline__ v16bf join16(v8bf lo, v8bf hi) {
  v16bf r;
#pragma unroll
  for (int i = 0; i < 8; ++i) { r[i] = lo[i]; r[i + 8] = hi[i]; }
  return r;
}

static __device__ __forceinline__ v8f vzero8() {
  v8f z;
#pragma unroll
  for (int i = 0; i < 8; ++i) z[i] = 0.0f;
  return z;
}

static __device__ __forceinline__ v8f wmma_bf16(v16bf a, v16bf b, v8f c) {
  // (neg_a, A, neg_b, B, c_mod, C, reuse_a, reuse_b)
  return __builtin_amdgcn_wmma_f32_16x16x32_bf16(false, a, false, b, (short)0, c,
                                                 false, false);
}

// ---------------- elementwise fp32 -> bf16 ----------------
__global__ void k_f32_to_bf16(const float* __restrict__ src,
                              bf16_t* __restrict__ dst, int n) {
  int i = blockIdx.x * blockDim.x + threadIdx.x;
  if (i < n) dst[i] = f2bf(src[i]);
}

// ---- pack weight [K,N] fp32 into per-(32x16)-tile WMMA B-fragment lane order ----
// Consumer: tile (kt,ntile) base = (kt*(N/16)+ntile)*512; lane L reads 16
// contiguous bf16 = column n = ntile*16 + (L&15), K = kt*32 + (L<16?0:8) + (j<8?j:j+8).
__global__ void k_pack_b(const float* __restrict__ w, bf16_t* __restrict__ pk,
                         int K, int N) {
  int idx = blockIdx.x * blockDim.x + threadIdx.x;
  if (idx >= K * N) return;
  int j      = idx & 15;
  int lane   = (idx >> 4) & 31;
  int tl     = idx >> 9;
  int ntiles = N >> 4;
  int ntile  = tl % ntiles;
  int kt     = tl / ntiles;
  int k = kt * 32 + ((lane < 16) ? 0 : 8) + ((j < 8) ? j : j + 8);
  int n = ntile * 16 + (lane & 15);
  pk[idx] = f2bf(w[(size_t)k * N + n]);
}

// ---------------- WMMA GEMM: C[32x64 per wave] = A(bf16 row-major) @ Bpack ----------------
// mode 0: QKV projection -> scatter to q (pre-scaled), k [B,H,T,64], vT [B,H,64,T]
// mode 1: out proj -> fp32 d_out
__global__ void __launch_bounds__(256)
k_gemm(const bf16_t* __restrict__ A, const bf16_t* __restrict__ Bp,
       const float* __restrict__ bias, int N, int K, int mode,
       float* __restrict__ outf, bf16_t* __restrict__ qb,
       bf16_t* __restrict__ kbuf, bf16_t* __restrict__ vtb) {
  const int wave = threadIdx.x >> 5;
  const int lane = threadIdx.x & 31;
  const int gw   = blockIdx.x * 8 + wave;
  const int ntilesW = N >> 6;                 // 64-wide wave tiles along N
  const int mt = (gw / ntilesW) << 5;
  const int nt = (gw % ntilesW) << 6;
  const int lr = lane & 15;
  const int kb16 = (lane < 16) ? 0 : 8;
  const int ntiles16 = N >> 4;

  v8f c[2][4];
#pragma unroll
  for (int i = 0; i < 2; ++i)
#pragma unroll
    for (int j = 0; j < 4; ++j) c[i][j] = vzero8();

  const bf16_t* arow0 = A + (size_t)(mt + lr) * K;
  const bf16_t* arow1 = arow0 + (size_t)16 * K;

  for (int kk = 0; kk < K; kk += 32) {
    v16bf a0 = join16(ld8(arow0 + kk + kb16), ld8(arow0 + kk + kb16 + 16));
    v16bf a1 = join16(ld8(arow1 + kk + kb16), ld8(arow1 + kk + kb16 + 16));
    const bf16_t* bp =
        Bp + (((size_t)(kk >> 5) * ntiles16 + (nt >> 4)) << 9) + lane * 16;
#pragma unroll
    for (int j = 0; j < 4; ++j) {
      v16bf bfrag = join16(ld8(bp), ld8(bp + 8));
      bp += 512;
      c[0][j] = wmma_bf16(a0, bfrag, c[0][j]);
      c[1][j] = wmma_bf16(a1, bfrag, c[1][j]);
    }
  }

#pragma unroll
  for (int i = 0; i < 2; ++i)
#pragma unroll
    for (int j = 0; j < 4; ++j)
#pragma unroll
      for (int v = 0; v < 8; ++v) {
        int rowM = mt + i * 16 + ((lane < 16) ? v : v + 8);
        int col  = nt + j * 16 + lr;
        float val = c[i][j][v] + bias[col];
        if (mode == 0) {
          int bidx = rowM >> 11;      // / TSEQ
          int t    = rowM & 2047;
          int s = col >> 10;          // 0:q 1:k 2:v
          int h = (col >> 6) & 15;
          int d = col & 63;
          size_t bh = (size_t)(bidx * NH + h);
          if (s == 0)
            qb[(bh * TSEQ + t) * HD + d] = f2bf(val * 0.125f);  // fold 1/sqrt(64)
          else if (s == 1)
            kbuf[(bh * TSEQ + t) * HD + d] = f2bf(val);
          else
            vtb[(bh * HD + d) * TSEQ + t] = f2bf(val);
        } else {
          outf[(size_t)rowM * N + col] = val;
        }
      }
}

// ---------------- causal flash attention: one wave = one 16-row query tile ----------------
__global__ void __launch_bounds__(256)
k_attn(const bf16_t* __restrict__ qb, const bf16_t* __restrict__ kbuf,
       const bf16_t* __restrict__ vtb, bf16_t* __restrict__ yh) {
  __shared__ float  Slds[8][16][33];  // padded: 33-bank stride, conflict-free
  __shared__ bf16_t Plds[8][16][40];  // 80B row stride: 16B-aligned vec loads
  __shared__ float  Crow[8][16];      // per-row corr / 1/l broadcast

  const int wave = threadIdx.x >> 5;
  const int lane = threadIdx.x & 31;
  const int gw   = blockIdx.x * 8 + wave;
  const int qt   = gw & 127;          // T/16 = 128 query tiles
  const int bh   = gw >> 7;           // B*H = 64
  const int lr   = lane & 15;
  const int kb16 = (lane < 16) ? 0 : 16;
  const int off16 = (lane < 16) ? 0 : 16;

  const bf16_t* Qp = qb   + (size_t)bh * TSEQ * HD;
  const bf16_t* Kp = kbuf + (size_t)bh * TSEQ * HD;
  const bf16_t* Vp = vtb  + (size_t)bh * HD * TSEQ;

  // Q A-fragments (resident for the whole loop), scale already folded in.
  const bf16_t* qrow = Qp + (size_t)(qt * 16 + lr) * HD;
  const v16bf aq0 = join16(ld8(qrow + kb16),      ld8(qrow + kb16 + 8));
  const v16bf aq1 = join16(ld8(qrow + 32 + kb16), ld8(qrow + 32 + kb16 + 8));

  v8f o[4];
#pragma unroll
  for (int f = 0; f < 4; ++f) o[f] = vzero8();
  float mrow = -__builtin_inff();
  float lrow = 0.0f;

  const int qlast = qt * 16 + 15;
  for (int jb = 0; jb <= qlast; jb += 32) {
    // ---- S = Q @ K^T for keys [jb, jb+32) : 4 WMMAs ----
    const bf16_t* kr0 = Kp + (size_t)(jb + lr) * HD + off16;
    const bf16_t* kr1 = kr0 + (size_t)16 * HD;
    v16bf b00 = join16(ld8(kr0),      ld8(kr0 + 8));
    v16bf b01 = join16(ld8(kr0 + 32), ld8(kr0 + 40));
    v8f s0 = wmma_bf16(aq0, b00, vzero8());
    s0 = wmma_bf16(aq1, b01, s0);
    v16bf b10 = join16(ld8(kr1),      ld8(kr1 + 8));
    v16bf b11 = join16(ld8(kr1 + 32), ld8(kr1 + 40));
    v8f s1 = wmma_bf16(aq0, b10, vzero8());
    s1 = wmma_bf16(aq1, b11, s1);

    // ---- C-fragment -> LDS (per-wave region; same-wave LDS ops are in order) ----
#pragma unroll
    for (int v = 0; v < 8; ++v) {
      int rm = (lane < 16) ? v : v + 8;
      Slds[wave][rm][lr]      = s0[v];
      Slds[wave][rm][lr + 16] = s1[v];
    }
    asm volatile("s_wait_dscnt 0x0" ::: "memory");

    // ---- online softmax row pass (lane = row) ----
    if (lane < 16) {
      const int qi  = qt * 16 + lane;
      const int lim = qi - jb;  // valid cols: j <= lim (causal)
      float mt = -__builtin_inff();
#pragma unroll
      for (int j = 0; j < 32; ++j)
        if (j <= lim) mt = fmaxf(mt, Slds[wave][lane][j]);
      float mnew = fmaxf(mrow, mt);
      float cf = (mnew == -__builtin_inff()) ? 1.0f : __expf(mrow - mnew);
      float pt = 0.0f;
#pragma unroll
      for (int j = 0; j < 32; ++j) {
        float p = 0.0f;
        if (j <= lim && mnew != -__builtin_inff())
          p = __expf(Slds[wave][lane][j] - mnew);
        Plds[wave][lane][j] = f2bf(p);
        pt += p;
      }
      lrow = lrow * cf + pt;
      mrow = mnew;
      Crow[wave][lane] = cf;
    }
    asm volatile("s_wait_dscnt 0x0" ::: "memory");

    // ---- rescale O accumulators by per-row correction ----
#pragma unroll
    for (int v = 0; v < 8; ++v) {
      int rm = (lane < 16) ? v : v + 8;
      float cf = Crow[wave][rm];
#pragma unroll
      for (int f = 0; f < 4; ++f) o[f][v] *= cf;
    }

    // ---- O += P @ V : rebuild P A-fragment from LDS, 4 WMMAs over d-tiles ----
    const bf16_t* prow = &Plds[wave][lr][0];
    v16bf ap = join16(ld8(prow + kb16), ld8(prow + kb16 + 8));
#pragma unroll
    for (int f = 0; f < 4; ++f) {
      const bf16_t* vr = Vp + (size_t)(f * 16 + lr) * TSEQ + jb + off16;
      v16bf bv = join16(ld8(vr), ld8(vr + 8));
      o[f] = wmma_bf16(ap, bv, o[f]);
    }
  }

  // ---- epilogue: divide by l, scatter to y [B,T,D] bf16 ----
  if (lane < 16) Crow[wave][lane] = 1.0f / lrow;
  asm volatile("s_wait_dscnt 0x0" ::: "memory");
  const int bidx = bh >> 4;   // / NH
  const int h    = bh & 15;
#pragma unroll
  for (int v = 0; v < 8; ++v) {
    int rm = (lane < 16) ? v : v + 8;
    float li = Crow[wave][rm];
    int t = qt * 16 + rm;
    bf16_t* yrow = yh + (size_t)(bidx * TSEQ + t) * DMODEL + h * HD;
#pragma unroll
    for (int f = 0; f < 4; ++f) yrow[f * 16 + lr] = f2bf(o[f][v] * li);
  }
}

extern "C" void kernel_launch(void* const* d_in, const int* in_sizes, int n_in,
                              void* d_out, int out_size, void* d_ws, size_t ws_size,
                              hipStream_t stream) {
  (void)in_sizes; (void)n_in; (void)out_size; (void)ws_size;
  const float* x      = (const float*)d_in[0];
  const float* w_qkv  = (const float*)d_in[1];
  const float* b_qkv  = (const float*)d_in[2];
  const float* w_proj = (const float*)d_in[3];
  const float* b_proj = (const float*)d_in[4];
  float* out = (float*)d_out;

  // workspace carve (all bf16): ~88 MB total
  char* ws = (char*)d_ws;
  const size_t szXh = (size_t)MTOT * DMODEL * 2;              // 16 MB
  const size_t szWq = (size_t)DMODEL * 3 * DMODEL * 2;        // 6 MB
  const size_t szWp = (size_t)DMODEL * DMODEL * 2;            // 2 MB
  const size_t szQ  = (size_t)BB * NH * TSEQ * HD * 2;        // 16 MB
  bf16_t* xh  = (bf16_t*)ws; ws += szXh;
  bf16_t* wqp = (bf16_t*)ws; ws += szWq;
  bf16_t* wpp = (bf16_t*)ws; ws += szWp;
  bf16_t* qb  = (bf16_t*)ws; ws += szQ;
  bf16_t* kb  = (bf16_t*)ws; ws += szQ;
  bf16_t* vt  = (bf16_t*)ws; ws += szQ;
  bf16_t* yh  = (bf16_t*)ws; ws += szXh;

  const int nx = MTOT * DMODEL;
  k_f32_to_bf16<<<(nx + 255) / 256, 256, 0, stream>>>(x, xh, nx);
  const int nwq = DMODEL * 3 * DMODEL;
  k_pack_b<<<(nwq + 255) / 256, 256, 0, stream>>>(w_qkv, wqp, DMODEL, 3 * DMODEL);
  const int nwp = DMODEL * DMODEL;
  k_pack_b<<<(nwp + 255) / 256, 256, 0, stream>>>(w_proj, wpp, DMODEL, DMODEL);

  // QKV projection: M=8192, N=3072, K=1024
  {
    int waves = (MTOT / 32) * ((3 * DMODEL) / 64);   // 12288
    k_gemm<<<waves / 8, 256, 0, stream>>>(xh, wqp, b_qkv, 3 * DMODEL, DMODEL, 0,
                                          nullptr, qb, kb, vt);
  }
  // flash attention: 64 (b,h) x 128 query tiles, 8 waves/block
  k_attn<<<(BB * NH * (TSEQ / 16)) / 8, 256, 0, stream>>>(qb, kb, vt, yh);
  // output projection: M=8192, N=1024, K=1024
  {
    int waves = (MTOT / 32) * (DMODEL / 64);         // 4096
    k_gemm<<<waves / 8, 256, 0, stream>>>(yh, wpp, b_proj, DMODEL, DMODEL, 1,
                                          out, nullptr, nullptr, nullptr);
  }
}
